// LSTM_42013370089785
// MI455X (gfx1250) — compile-verified, matrix operand order checked by reading, not run on hardware
//
#include <hip/hip_runtime.h>

typedef __attribute__((ext_vector_type(16))) __bf16 v16bf;
typedef __attribute__((ext_vector_type(8)))  __bf16 v8bf;
typedef __attribute__((ext_vector_type(8)))  float  v8f;

#define B_SZ 256
#define T_SZ 512
#define I_SZ 512
#define H_SZ 1024
#define K_SZ 1536            // I + H (combined GEMM K)
#define G4_SZ 4096           // 4 * H

// Tiling: per workgroup 64 batch-rows x 64 within-gate cols x 4 gates.
#define BM 64
#define BJ 64
#define KC 64                // K chunk per LDS stage (2 WMMA k-steps)
#define KITERS (K_SZ / KC)   // 24

__device__ __forceinline__ float sigmoidf_(float x) {
  return 1.0f / (1.0f + __expf(-x));
}

union BF16x16 { v16bf v; v8bf h[2]; };

// 128-byte row: 8 async b128 copies, LDS[lds+off] = MEM[g+off] (ASYNCcnt).
__device__ __forceinline__ void async_row128(unsigned lds, const __bf16* g) {
  asm volatile("global_load_async_to_lds_b128 %0, %1, off"            :: "v"(lds), "v"(g) : "memory");
  asm volatile("global_load_async_to_lds_b128 %0, %1, off offset:16"  :: "v"(lds), "v"(g) : "memory");
  asm volatile("global_load_async_to_lds_b128 %0, %1, off offset:32"  :: "v"(lds), "v"(g) : "memory");
  asm volatile("global_load_async_to_lds_b128 %0, %1, off offset:48"  :: "v"(lds), "v"(g) : "memory");
  asm volatile("global_load_async_to_lds_b128 %0, %1, off offset:64"  :: "v"(lds), "v"(g) : "memory");
  asm volatile("global_load_async_to_lds_b128 %0, %1, off offset:80"  :: "v"(lds), "v"(g) : "memory");
  asm volatile("global_load_async_to_lds_b128 %0, %1, off offset:96"  :: "v"(lds), "v"(g) : "memory");
  asm volatile("global_load_async_to_lds_b128 %0, %1, off offset:112" :: "v"(lds), "v"(g) : "memory");
}

// ---------------------------------------------------------------------------
// Pack the 8 gate weight matrices into W^T bf16: wT[n][k], n = gate*1024 + j,
// k in [0,1536): k<512 -> Wx rows (w*2), k>=512 -> Wh rows (w*1).
// ---------------------------------------------------------------------------
__global__ __launch_bounds__(256) void lstm_pack_w(
    __bf16* __restrict__ wT,
    const float* __restrict__ wlr1, const float* __restrict__ wlr2,
    const float* __restrict__ wpr1, const float* __restrict__ wpr2,
    const float* __restrict__ wp1,  const float* __restrict__ wp2,
    const float* __restrict__ wo1,  const float* __restrict__ wo2)
{
  int idx = blockIdx.x * 256 + threadIdx.x;
  if (idx >= G4_SZ * K_SZ) return;
  int n = idx / K_SZ, k = idx - n * K_SZ;
  int g = n >> 10, j = n & 1023;
  const float* wx; const float* wh;
  switch (g) {
    case 0:  wx = wlr2; wh = wlr1; break;
    case 1:  wx = wpr2; wh = wpr1; break;
    case 2:  wx = wp2;  wh = wp1;  break;
    default: wx = wo2;  wh = wo1;  break;
  }
  float v = (k < I_SZ) ? wx[(size_t)k * H_SZ + j]
                       : wh[(size_t)(k - I_SZ) * H_SZ + j];
  wT[idx] = (__bf16)v;
}

// ---------------------------------------------------------------------------
// Zero c / h buffers, concatenate biases.
// ---------------------------------------------------------------------------
__global__ __launch_bounds__(256) void lstm_init(
    float* __restrict__ c, float* __restrict__ h0, float* __restrict__ h1,
    float* __restrict__ bias4,
    const float* __restrict__ b0, const float* __restrict__ b1,
    const float* __restrict__ b2, const float* __restrict__ b3)
{
  int idx = blockIdx.x * 256 + threadIdx.x;
  if (idx < B_SZ * H_SZ) { c[idx] = 0.f; h0[idx] = 0.f; h1[idx] = 0.f; }
  if (idx < H_SZ) {
    bias4[idx]             = b0[idx];
    bias4[H_SZ + idx]      = b1[idx];
    bias4[2 * H_SZ + idx]  = b2[idx];
    bias4[3 * H_SZ + idx]  = b3[idx];
  }
}

// ---------------------------------------------------------------------------
// One recurrence step: G = [x_t | h_prev] @ [Wx;Wh] + b for 4 gates, fused
// with the elementwise LSTM cell update. Grid: (256/BM, 1024/BJ) WGs of 256.
// Double-buffered LDS; B tiles via async-to-LDS, A staged fp32->bf16.
// ---------------------------------------------------------------------------
__global__ __launch_bounds__(256) void lstm_step(
    const float* __restrict__ x, const __bf16* __restrict__ wT,
    const float* __restrict__ bias4,
    const float* __restrict__ hin, float* __restrict__ hout,
    float* __restrict__ cst, int t)
{
  __shared__ __align__(16) __bf16 As[2][BM][KC];       // 2 x 8 KB
  __shared__ __align__(16) __bf16 Bs[2][4 * BJ][KC];   // 2 x 32 KB

  const int tid  = threadIdx.x;
  const int lane = tid & 31;
  const int wave = tid >> 5;
  const int mp   = wave >> 2;        // 0..1 : pair of 16-row subtiles (P=2)
  const int ji   = wave & 3;         // 0..3 : 16-col subtile within BJ
  const int m16  = lane & 15;
  const int hi   = lane >> 4;

  const int mBase = blockIdx.x * BM; // batch rows
  const int jBase = blockIdx.y * BJ; // within-gate cols

  v8f acc[2][4] = {};                // [m-subtile][gate]

  // Staging assignments (all 256 threads):
  // B: 256 rows (4 gates x 64 cols), one 128B row per thread, async-to-LDS.
  const int bg = tid >> 6, bjc = tid & 63;
  const __bf16* bRowPtr = wT + (size_t)(bg * H_SZ + jBase + bjc) * K_SZ;
  // A: 64 rows x 4 segments of 16 halfs, fp32->bf16 convert through VGPRs.
  const int arow = tid >> 2, aseg = tid & 3;
  const int ab   = mBase + arow;

  auto a_src = [&](int kb) -> const float* {
    const int kk = kb + aseg * 16;
    return (kk < I_SZ) ? (x + ((size_t)ab * T_SZ + t) * I_SZ + kk)
                       : (hin + (size_t)ab * H_SZ + (kk - I_SZ));
  };

  // ---- prologue: stage chunk 0 into buffer 0 ----
  async_row128((unsigned)(size_t)&Bs[0][tid][0], bRowPtr);
  {
    const float* src = a_src(0);
    float4 f0 = *(const float4*)(src + 0);
    float4 f1 = *(const float4*)(src + 4);
    float4 f2 = *(const float4*)(src + 8);
    float4 f3 = *(const float4*)(src + 12);
    v8bf lo = { (__bf16)f0.x, (__bf16)f0.y, (__bf16)f0.z, (__bf16)f0.w,
                (__bf16)f1.x, (__bf16)f1.y, (__bf16)f1.z, (__bf16)f1.w };
    v8bf hi8 = { (__bf16)f2.x, (__bf16)f2.y, (__bf16)f2.z, (__bf16)f2.w,
                 (__bf16)f3.x, (__bf16)f3.y, (__bf16)f3.z, (__bf16)f3.w };
    *(v8bf*)&As[0][arow][aseg * 16]     = lo;
    *(v8bf*)&As[0][arow][aseg * 16 + 8] = hi8;
  }
  asm volatile("s_wait_asynccnt 0" ::: "memory");
  __syncthreads();

  int buf = 0;
  for (int it = 0; it < KITERS; ++it) {
    const bool pf = (it + 1 < KITERS);
    float4 f0, f1, f2, f3;
    if (pf) {
      // prefetch next chunk: async B into buf^1, A globals into VGPRs
      async_row128((unsigned)(size_t)&Bs[buf ^ 1][tid][0],
                   bRowPtr + (it + 1) * KC);
      const float* src = a_src((it + 1) * KC);
      f0 = *(const float4*)(src + 0);
      f1 = *(const float4*)(src + 4);
      f2 = *(const float4*)(src + 8);
      f3 = *(const float4*)(src + 12);
    }

    // ---- compute on current buffer: 16 WMMAs per wave ----
    #pragma unroll
    for (int kc = 0; kc < 2; ++kc) {
      BF16x16 ua[2];
      #pragma unroll
      for (int p = 0; p < 2; ++p) {
        const __bf16* ap = &As[buf][(mp * 2 + p) * 16 + m16][kc * 32 + hi * 8];
        ua[p].h[0] = *(const v8bf*)ap;
        ua[p].h[1] = *(const v8bf*)(ap + 16);
      }
      #pragma unroll
      for (int g = 0; g < 4; ++g) {
        BF16x16 ub;   // B fragment, reused by both A rows
        const __bf16* bp = &Bs[buf][g * BJ + ji * 16 + m16][kc * 32 + hi * 8];
        ub.h[0] = *(const v8bf*)bp;
        ub.h[1] = *(const v8bf*)(bp + 16);
        acc[0][g] = __builtin_amdgcn_wmma_f32_16x16x32_bf16(
            false, ua[0].v, false, ub.v, (short)0, acc[0][g], false, false);
        acc[1][g] = __builtin_amdgcn_wmma_f32_16x16x32_bf16(
            false, ua[1].v, false, ub.v, (short)0, acc[1][g], false, false);
      }
    }

    if (pf) {
      // finish A staging into buf^1 after compute (loads overlapped above)
      v8bf lo = { (__bf16)f0.x, (__bf16)f0.y, (__bf16)f0.z, (__bf16)f0.w,
                  (__bf16)f1.x, (__bf16)f1.y, (__bf16)f1.z, (__bf16)f1.w };
      v8bf hi8 = { (__bf16)f2.x, (__bf16)f2.y, (__bf16)f2.z, (__bf16)f2.w,
                   (__bf16)f3.x, (__bf16)f3.y, (__bf16)f3.z, (__bf16)f3.w };
      *(v8bf*)&As[buf ^ 1][arow][aseg * 16]     = lo;
      *(v8bf*)&As[buf ^ 1][arow][aseg * 16 + 8] = hi8;
    }
    asm volatile("s_wait_asynccnt 0" ::: "memory");
    __syncthreads();
    buf ^= 1;
  }

  // Fused gate activations + cell update. C-layout: VGPR r -> row r (+8 for
  // hi lanes), column = lane&15. Each (b,j) owned by exactly one lane slot.
  const int j  = jBase + ji * 16 + m16;
  const float bF = bias4[j];
  const float bI = bias4[H_SZ + j];
  const float bP = bias4[2 * H_SZ + j];
  const float bO = bias4[3 * H_SZ + j];
  #pragma unroll
  for (int p = 0; p < 2; ++p) {
    #pragma unroll
    for (int r = 0; r < 8; ++r) {
      const int b    = mBase + (mp * 2 + p) * 16 + r + hi * 8;
      const size_t o = (size_t)b * H_SZ + j;
      float f  = sigmoidf_(acc[p][0][r] + bF);
      float ig = sigmoidf_(acc[p][1][r] + bI);
      float pg = tanhf(acc[p][2][r] + bP);
      float og = sigmoidf_(acc[p][3][r] + bO);
      float cn = cst[o] * f + ig * pg;
      cst[o]  = cn;
      hout[o] = og * tanhf(cn);
    }
  }
}

// ---------------------------------------------------------------------------
// Final FC: out[b][o] = h[b] . fc_w[o] + fc_b[o]  (256x128, K=1024, fp32)
// ---------------------------------------------------------------------------
__global__ __launch_bounds__(256) void lstm_fc(
    const float* __restrict__ h, const float* __restrict__ fcw,
    const float* __restrict__ fcb, float* __restrict__ out)
{
  int idx = blockIdx.x * 256 + threadIdx.x;  // 32768 outputs
  int b = idx >> 7, o = idx & 127;
  const float* hr = h + (size_t)b * H_SZ;
  const float* wr = fcw + (size_t)o * H_SZ;
  float acc = 0.f;
  #pragma unroll 4
  for (int k = 0; k < H_SZ; k += 4) {
    float4 hv = *(const float4*)(hr + k);
    float4 wv = *(const float4*)(wr + k);
    acc += hv.x * wv.x + hv.y * wv.y + hv.z * wv.z + hv.w * wv.w;
  }
  out[idx] = acc + fcb[o];
}

// ---------------------------------------------------------------------------
extern "C" void kernel_launch(void* const* d_in, const int* in_sizes, int n_in,
                              void* d_out, int out_size, void* d_ws, size_t ws_size,
                              hipStream_t stream)
{
  const float* x    = (const float*)d_in[0];
  const float* wlr1 = (const float*)d_in[1];
  const float* wlr2 = (const float*)d_in[2];
  const float* blr1 = (const float*)d_in[3];
  const float* wpr1 = (const float*)d_in[4];
  const float* wpr2 = (const float*)d_in[5];
  const float* bpr1 = (const float*)d_in[6];
  const float* wp1  = (const float*)d_in[7];
  const float* wp2  = (const float*)d_in[8];
  const float* bp1  = (const float*)d_in[9];
  const float* wo1  = (const float*)d_in[10];
  const float* wo2  = (const float*)d_in[11];
  const float* bo1  = (const float*)d_in[12];
  const float* fcw  = (const float*)d_in[13];
  const float* fcb  = (const float*)d_in[14];

  char* ws = (char*)d_ws;
  const size_t OFF_WT   = 0;                                   // 12,582,912 B
  const size_t OFF_BIAS = OFF_WT + (size_t)G4_SZ * K_SZ * 2;
  const size_t OFF_C    = OFF_BIAS + (size_t)G4_SZ * 4;
  const size_t OFF_H0   = OFF_C + (size_t)B_SZ * H_SZ * 4;
  const size_t OFF_H1   = OFF_H0 + (size_t)B_SZ * H_SZ * 4;    // end ~16.8 MB
  __bf16* wT   = (__bf16*)(ws + OFF_WT);
  float* bias4 = (float*)(ws + OFF_BIAS);
  float* cst   = (float*)(ws + OFF_C);
  float* h0    = (float*)(ws + OFF_H0);
  float* h1    = (float*)(ws + OFF_H1);

  lstm_pack_w<<<(G4_SZ * K_SZ + 255) / 256, 256, 0, stream>>>(
      wT, wlr1, wlr2, wpr1, wpr2, wp1, wp2, wo1, wo2);
  lstm_init<<<(B_SZ * H_SZ + 255) / 256, 256, 0, stream>>>(
      cst, h0, h1, bias4, blr1, bpr1, bp1, bo1);

  for (int t = 0; t < T_SZ; ++t) {
    const float* hin = (t & 1) ? h1 : h0;   // ping-pong h to avoid races
    float*       hqo = (t & 1) ? h0 : h1;
    lstm_step<<<dim3(B_SZ / BM, H_SZ / BJ), 256, 0, stream>>>(
        x, wT, bias4, hin, hqo, cst, t);
  }
  // T=512 (even) -> final h lands in h0.
  lstm_fc<<<(B_SZ * 128 + 255) / 256, 256, 0, stream>>>(h0, fcw, fcb,
                                                        (float*)d_out);
  (void)in_sizes; (void)n_in; (void)out_size; (void)ws_size;
}
